// MultiHeadAttention_68478958567570
// MI455X (gfx1250) — matrix operand order repaired and checked by reading, not compiled, not run on hardware
//
#include <hip/hip_runtime.h>
#include <stdint.h>

#define B_   4
#define S_   2048
#define HID  768
#define NH   12
#define HD   64
#define NQKV 2304
#define TOK  (B_ * S_)   // 8192

typedef __attribute__((ext_vector_type(16))) __bf16 v16bf;
typedef __attribute__((ext_vector_type(8)))  float  v8f;

#if defined(__gfx1250__) &&                                              \
    __has_builtin(__builtin_amdgcn_global_load_async_to_lds_b128) &&     \
    __has_builtin(__builtin_amdgcn_s_wait_asynccnt)
#define ASYNC_LDS 1
typedef int v4i_ __attribute__((ext_vector_type(4)));
typedef __attribute__((address_space(1))) v4i_* gas_p;   // global int4*
typedef __attribute__((address_space(3))) v4i_* las_p;   // LDS int4*
#else
#define ASYNC_LDS 0
#endif

struct U32x8 { uint4 lo, hi; };

static __device__ __forceinline__ v16bf pack16(uint4 lo, uint4 hi) {
  U32x8 t; t.lo = lo; t.hi = hi;
  return __builtin_bit_cast(v16bf, t);
}

static __device__ __forceinline__ uint16_t f2bf(float f) {
  uint32_t u = __builtin_bit_cast(uint32_t, f);
  u += 0x7FFFu + ((u >> 16) & 1u);   // round-to-nearest-even
  return (uint16_t)(u >> 16);
}

static __device__ __forceinline__ uint4 ld4(const uint16_t* p) {
  return *reinterpret_cast<const uint4*>(p);
}

static __device__ __forceinline__ v8f vzero8() {
  v8f z = {0.0f, 0.0f, 0.0f, 0.0f, 0.0f, 0.0f, 0.0f, 0.0f};
  return z;
}

static __device__ __forceinline__ float rmax16(float v) {
  #pragma unroll
  for (int m = 1; m < 16; m <<= 1) v = fmaxf(v, __shfl_xor(v, m, 32));
  return v;
}
static __device__ __forceinline__ float rsum16(float v) {
  #pragma unroll
  for (int m = 1; m < 16; m <<= 1) v += __shfl_xor(v, m, 32);
  return v;
}

static __device__ __forceinline__ v8f wmma_bf16(v16bf a, v16bf b, v8f c) {
  return __builtin_amdgcn_wmma_f32_16x16x32_bf16(false, a, false, b,
                                                 (short)0, c, false, false);
}

// ---------------------------------------------------------------------------
// fp32 -> bf16 conversion kernels
// ---------------------------------------------------------------------------
__global__ void cvt_bf16_kernel(const float* __restrict__ s,
                                uint16_t* __restrict__ d, int n) {
  int i = blockIdx.x * 256 + threadIdx.x;
  if (i < n) d[i] = f2bf(s[i]);
}

// s is [K][N] row-major; d becomes [N][K] bf16 (K contiguous)
__global__ void cvt_bf16_T_kernel(const float* __restrict__ s,
                                  uint16_t* __restrict__ d, int K, int N) {
  int i = blockIdx.x * 256 + threadIdx.x;
  if (i < N * K) {
    int n = i / K;
    int k = i - n * K;
    d[i] = f2bf(s[(size_t)k * N + n]);
  }
}

// ---------------------------------------------------------------------------
// bf16 WMMA GEMM: C[M][N] = A[M][K] * BT[N][K]^T + bias
//   mode 0: QKV epilogue -> scatter to Q[B,H,S,D] (x0.125), K[B,H,S,D],
//           Vt[B,H,D,S] as bf16
//   mode 1: proj epilogue -> fp32 out [M][N]
// Block tile 128x128x32, 256 threads (8 waves), wave tile 32x64.
// Double-buffered LDS; tile staging via async global->LDS when available.
// ---------------------------------------------------------------------------
__global__ __launch_bounds__(256) void gemm_bf16_wmma(
    const uint16_t* __restrict__ A, const uint16_t* __restrict__ BT,
    const float* __restrict__ bias, int M, int N, int K, int mode,
    uint16_t* __restrict__ q_out, uint16_t* __restrict__ k_out,
    uint16_t* __restrict__ vT_out, float* __restrict__ f_out) {
  __shared__ __align__(16) uint16_t lA[2][128 * 32];
  __shared__ __align__(16) uint16_t lB[2][128 * 32];

  const int tid  = threadIdx.x;
  const int lane = tid & 31;
  const int wave = tid >> 5;
  const int half = lane >> 4;
  const int l16  = lane & 15;
  const int wm   = wave & 3;   // wave row (4 x 32 = 128)
  const int wn   = wave >> 2;  // wave col (2 x 64 = 128)
  const int rowBase = blockIdx.x * 128;
  const int colBase = blockIdx.y * 128;

  // stage one 128x32 bf16 tile pair into LDS buffer `buf`
  auto copy_tile = [&](int kt, int buf) {
    #pragma unroll
    for (int t = 0; t < 2; t++) {
      int c  = tid + t * 256;   // 0..511
      int r  = c >> 2;          // 0..127
      int ko = (c & 3) * 8;     // 0,8,16,24
      const uint16_t* ga = &A[(size_t)(rowBase + r) * K + kt + ko];
      const uint16_t* gb = &BT[(size_t)(colBase + r) * K + kt + ko];
#if ASYNC_LDS
      __builtin_amdgcn_global_load_async_to_lds_b128(
          (gas_p)ga, (las_p)&lA[buf][r * 32 + ko], 0, 0);
      __builtin_amdgcn_global_load_async_to_lds_b128(
          (gas_p)gb, (las_p)&lB[buf][r * 32 + ko], 0, 0);
#else
      *reinterpret_cast<uint4*>(&lA[buf][r * 32 + ko]) = ld4(ga);
      *reinterpret_cast<uint4*>(&lB[buf][r * 32 + ko]) = ld4(gb);
#endif
    }
  };

  v8f acc[2][4];
  #pragma unroll
  for (int i = 0; i < 2; i++)
    #pragma unroll
    for (int j = 0; j < 4; j++) acc[i][j] = vzero8();

  int cur = 0;
  copy_tile(0, 0);
#if ASYNC_LDS
  __builtin_amdgcn_s_wait_asynccnt(0);
#endif
  __syncthreads();

  for (int kt = 0; kt < K; kt += 32) {
    if (kt + 32 < K) copy_tile(kt + 32, cur ^ 1);

    v16bf av[2], bv[4];
    #pragma unroll
    for (int mi = 0; mi < 2; mi++) {
      // A layout: lane=m, elems 0..7 -> k=half*8+i, elems 8..15 -> k=16+half*8+i
      const uint16_t* p = &lA[cur][(wm * 32 + mi * 16 + l16) * 32];
      av[mi] = pack16(ld4(p + half * 8), ld4(p + 16 + half * 8));
    }
    #pragma unroll
    for (int ni = 0; ni < 4; ni++) {
      // B layout: lane=n, elems 0..7 -> k=half*8+i, elems 8..15 -> k=16+half*8+i
      const uint16_t* p = &lB[cur][(wn * 64 + ni * 16 + l16) * 32];
      bv[ni] = pack16(ld4(p + half * 8), ld4(p + 16 + half * 8));
    }
    #pragma unroll
    for (int mi = 0; mi < 2; mi++)
      #pragma unroll
      for (int ni = 0; ni < 4; ni++)
        acc[mi][ni] = wmma_bf16(av[mi], bv[ni], acc[mi][ni]);

#if ASYNC_LDS
    __builtin_amdgcn_s_wait_asynccnt(0);
#endif
    __syncthreads();
    cur ^= 1;
  }

  // Epilogue. C/D layout: lane -> n = l16; VGPR r -> m = half*8 + r.
  #pragma unroll
  for (int mi = 0; mi < 2; mi++) {
    #pragma unroll
    for (int ni = 0; ni < 4; ni++) {
      int col = colBase + wn * 64 + ni * 16 + l16;
      float bias_c = bias[col];
      if (mode == 1) {
        #pragma unroll
        for (int r = 0; r < 8; r++) {
          int row = rowBase + wm * 32 + mi * 16 + half * 8 + r;
          f_out[(size_t)row * N + col] = acc[mi][ni][r] + bias_c;
        }
      } else {
        int which = col / HID;          // 0=q 1=k 2=v
        int rem   = col - which * HID;
        int h     = rem >> 6;
        int d     = rem & 63;
        #pragma unroll
        for (int r = 0; r < 8; r++) {
          int row = rowBase + wm * 32 + mi * 16 + half * 8 + r;  // token
          int b   = row >> 11;
          int s   = row & 2047;
          float v = acc[mi][ni][r] + bias_c;
          if (which == 0) {
            q_out[((size_t)(b * NH + h) * S_ + s) * HD + d] = f2bf(v * 0.125f);
          } else if (which == 1) {
            k_out[((size_t)(b * NH + h) * S_ + s) * HD + d] = f2bf(v);
          } else {
            vT_out[((size_t)(b * NH + h) * HD + d) * S_ + s] = f2bf(v);
          }
        }
      }
    }
  }
}

// ---------------------------------------------------------------------------
// Fused flash attention. Grid: B*NH*(S/64) blocks of 128 threads (4 waves).
// Each wave owns 16 query rows; iterates keys in blocks of 32 with online
// softmax; Q pre-scaled by 1/sqrt(64). Output bf16 [B,S,HID].
// ---------------------------------------------------------------------------
__global__ __launch_bounds__(128) void attn_wmma(
    const uint16_t* __restrict__ Q,   // [B,H,S,D]
    const uint16_t* __restrict__ Kc,  // [B,H,S,D]
    const uint16_t* __restrict__ Vt,  // [B,H,D,S]
    uint16_t* __restrict__ O) {       // [B,S,HID] bf16
  __shared__ __align__(16) uint16_t pbuf[4][16 * 32];  // wave-private P tiles

  const int lane = threadIdx.x & 31;
  const int wave = threadIdx.x >> 5;
  const int half = lane >> 4;
  const int l16  = lane & 15;

  int bid        = blockIdx.x;
  const int qblk = bid & 31;   // S/64 = 32
  bid >>= 5;
  const int h = bid % NH;
  const int b = bid / NH;
  const int qbase = qblk * 64 + wave * 16;

  const uint16_t* qp = Q  + (size_t)(b * NH + h) * S_ * HD;
  const uint16_t* kp = Kc + (size_t)(b * NH + h) * S_ * HD;
  const uint16_t* vp = Vt + (size_t)(b * NH + h) * HD * S_;

  // Q A-operands: two k-chunks of 32 over D=64
  v16bf aq[2];
  {
    const uint16_t* qrow = qp + (size_t)(qbase + l16) * HD;
    #pragma unroll
    for (int c = 0; c < 2; c++)
      aq[c] = pack16(ld4(qrow + c * 32 + half * 8),
                     ld4(qrow + c * 32 + 16 + half * 8));
  }

  v8f ov[4];
  #pragma unroll
  for (int ni = 0; ni < 4; ni++) ov[ni] = vzero8();
  float mrow[8], lrow[8];
  #pragma unroll
  for (int r = 0; r < 8; r++) { mrow[r] = -3.0e38f; lrow[r] = 0.0f; }

  for (int kb = 0; kb < S_; kb += 32) {
    // prefetch next key/value block (K/V live in L2; hide hit latency)
    if (kb + 32 < S_) {
      __builtin_prefetch(kp + (size_t)(kb + 32 + l16) * HD, 0, 0);
      __builtin_prefetch(vp + (size_t)l16 * S_ + kb + 32, 0, 0);
    }

    // scores: two 16-key tiles
    v8f sc[2];
    #pragma unroll
    for (int t = 0; t < 2; t++) {
      const uint16_t* krow = kp + (size_t)(kb + t * 16 + l16) * HD;
      v16bf b0 = pack16(ld4(krow + half * 8), ld4(krow + 16 + half * 8));
      v16bf b1 = pack16(ld4(krow + 32 + half * 8), ld4(krow + 32 + 16 + half * 8));
      v8f z = vzero8();
      z = wmma_bf16(aq[0], b0, z);
      z = wmma_bf16(aq[1], b1, z);
      sc[t] = z;
    }

    // online softmax across these 32 keys (rows live across 16-lane groups)
    #pragma unroll
    for (int r = 0; r < 8; r++) {
      float vm = rmax16(fmaxf(sc[0][r], sc[1][r]));
      float nm = fmaxf(mrow[r], vm);
      float alpha = __expf(mrow[r] - nm);
      mrow[r] = nm;
      float p0 = __expf(sc[0][r] - nm);
      float p1 = __expf(sc[1][r] - nm);
      lrow[r] = lrow[r] * alpha + rsum16(p0 + p1);
      #pragma unroll
      for (int ni = 0; ni < 4; ni++) ov[ni][r] *= alpha;
      int rowm = half * 8 + r;
      pbuf[wave][rowm * 32 + l16]      = f2bf(p0);
      pbuf[wave][rowm * 32 + 16 + l16] = f2bf(p1);
    }

    // P (C-layout) -> A-operand via wave-private LDS
    const uint16_t* pr = &pbuf[wave][l16 * 32];
    v16bf pa = pack16(ld4(pr + half * 8), ld4(pr + 16 + half * 8));

    // O += P * V  (V stored transposed: per-lane d row, keys contiguous)
    #pragma unroll
    for (int ni = 0; ni < 4; ni++) {
      const uint16_t* vrow = vp + (size_t)(ni * 16 + l16) * S_ + kb;
      v16bf bv = pack16(ld4(vrow + half * 8), ld4(vrow + 16 + half * 8));
      ov[ni] = wmma_bf16(pa, bv, ov[ni]);
    }
  }

  // normalize + store bf16 [B,S,HID]
  #pragma unroll
  for (int r = 0; r < 8; r++) {
    int rowm  = half * 8 + r;
    float inv = 1.0f / lrow[r];
    #pragma unroll
    for (int ni = 0; ni < 4; ni++) {
      int col = h * HD + ni * 16 + l16;
      O[(size_t)(b * S_ + qbase + rowm) * HID + col] = f2bf(ov[ni][r] * inv);
    }
  }
}

// ---------------------------------------------------------------------------
extern "C" void kernel_launch(void* const* d_in, const int* in_sizes, int n_in,
                              void* d_out, int out_size, void* d_ws,
                              size_t ws_size, hipStream_t stream) {
  const float* x      = (const float*)d_in[0];
  const float* qkv_w  = (const float*)d_in[1];
  const float* qkv_b  = (const float*)d_in[2];
  const float* proj_w = (const float*)d_in[3];
  const float* proj_b = (const float*)d_in[4];
  float* out = (float*)d_out;

  char* ws = (char*)d_ws;
  size_t off = 0;
  auto carve = [&](size_t bytes) -> void* {
    void* p = ws + off;
    off += (bytes + 255) & ~(size_t)255;
    return p;
  };
  uint16_t* xb     = (uint16_t*)carve((size_t)TOK * HID * 2);
  uint16_t* wqkvT  = (uint16_t*)carve((size_t)NQKV * HID * 2);
  uint16_t* wprojT = (uint16_t*)carve((size_t)HID * HID * 2);
  uint16_t* qb     = (uint16_t*)carve((size_t)TOK * HID * 2);
  uint16_t* kbuf   = (uint16_t*)carve((size_t)TOK * HID * 2);
  uint16_t* vtb    = (uint16_t*)carve((size_t)TOK * HID * 2);
  uint16_t* attnb  = (uint16_t*)carve((size_t)TOK * HID * 2);

  cvt_bf16_kernel<<<(TOK * HID + 255) / 256, 256, 0, stream>>>(x, xb, TOK * HID);
  cvt_bf16_T_kernel<<<(HID * NQKV + 255) / 256, 256, 0, stream>>>(qkv_w, wqkvT,
                                                                  HID, NQKV);
  cvt_bf16_T_kernel<<<(HID * HID + 255) / 256, 256, 0, stream>>>(proj_w, wprojT,
                                                                 HID, HID);

  gemm_bf16_wmma<<<dim3(TOK / 128, NQKV / 128), 256, 0, stream>>>(
      xb, wqkvT, qkv_b, TOK, NQKV, HID, /*mode=*/0, qb, kbuf, vtb, nullptr);

  attn_wmma<<<B_ * NH * (S_ / 64), 128, 0, stream>>>(qb, kbuf, vtb, attnb);

  gemm_bf16_wmma<<<dim3(TOK / 128, HID / 128), 256, 0, stream>>>(
      attnb, wprojT, proj_b, TOK, HID, HID, /*mode=*/1, nullptr, nullptr,
      nullptr, out);
}
